// NBSampler_24816321036782
// MI455X (gfx1250) — matrix operand order repaired and checked by reading, not compiled, not run on hardware
//
#include <hip/hip_runtime.h>
#include <hip/hip_bf16.h>

// ---------------------------------------------------------------------------
// Problem constants (match reference): B=256 rows, D=4096 dims, MAX_R=31 steps
// ---------------------------------------------------------------------------
#define B_DIM 256
#define D_DIM 4096
#define MAXR  31

typedef __attribute__((ext_vector_type(16))) __bf16 v16bf;
typedef __attribute__((ext_vector_type(8)))  __bf16 v8bf;
typedef __attribute__((ext_vector_type(4)))  __bf16 v4bf;
typedef __attribute__((ext_vector_type(8)))  float  v8f;
typedef __attribute__((ext_vector_type(4)))  unsigned v4u;
typedef __attribute__((ext_vector_type(8)))  int      v8i;
typedef __attribute__((ext_vector_type(4)))  int      v4i;

#if __has_builtin(__builtin_amdgcn_tensor_load_to_lds) && \
    __has_builtin(__builtin_amdgcn_s_wait_tensorcnt)
#define USE_TDM 1
#else
#define USE_TDM 0
#endif

// ---------------------------------------------------------------------------
// Small deterministic RNG (PCG-style hash) + helpers
// ---------------------------------------------------------------------------
__device__ __forceinline__ unsigned pcg_hash(unsigned v) {
    v = v * 747796405u + 2891336453u;
    unsigned w = ((v >> ((v >> 28) + 4u)) ^ v) * 277803737u;
    return (w >> 22) ^ w;
}
__device__ __forceinline__ float u01(unsigned h) {
    return (float)(h >> 8) * (1.0f / 16777216.0f) + (0.5f / 16777216.0f);
}

#define NEG_BIG (-3.402823466e38f)

// ---------------------------------------------------------------------------
// Kernel 1: convert W (f32, DxD) -> bf16 once (float4 loads, 8B stores).
// 32 MB result stays L2-resident for both GEMM passes (192 MB L2).
// ---------------------------------------------------------------------------
__global__ __launch_bounds__(256) void convert_w_kernel(
    const float* __restrict__ W, __bf16* __restrict__ Wb)
{
    const size_t base = (size_t)blockIdx.x * D_DIM;
    // 4096 floats per row = 1024 float4; 256 threads -> 4 float4 each
    for (int e = 0; e < 4; ++e) {
        size_t v = (size_t)e * 256 + threadIdx.x;       // float4 index in row
        float4 f = *(const float4*)&W[base + v * 4];
        v4bf pk = { (__bf16)f.x, (__bf16)f.y, (__bf16)f.z, (__bf16)f.w };
        *(v4bf*)&Wb[base + v * 4] = pk;
    }
}

// ---------------------------------------------------------------------------
// Kernel 2: grad[B,D] = Xbits[B,D] @ Wbf16[D,D] + bias  via v_wmma_f32_16x16x32_bf16
//   - block tile M=128, N=64, K-stage=64 (2 WMMA substeps per barrier)
//   - W is symmetric => B tile W[k,n] == W[n,k]: transposed B tile is a set of
//     contiguous row segments of W -> TDM-friendly 2D tile (or wide copies).
//   - B tile staged by TENSOR_LOAD_TO_LDS (wave 0 issues, s_wait_tensorcnt),
//     with TDM pad fields writing the +8 bf16 bank-conflict padding.
//   - A tile staged with float4 loads + packed f32->bf16 + 8B LDS stores.
// ---------------------------------------------------------------------------
constexpr int BM = 128, BN = 64, BK = 64, LDP = 8;
constexpr unsigned ABYTES = BM * (BK + LDP) * 2;   // LDS offset of BsT = 18432

struct SMem { __bf16 As[BM][BK + LDP]; __bf16 BsT[BN][BK + LDP]; };

__global__ __launch_bounds__(256) void gemm_wmma_kernel(
    const float*  __restrict__ X,      // [B,D] 0/1 bits as f32
    const __bf16* __restrict__ Wb,     // [D,D] bf16, row-major (symmetric)
    const float*  __restrict__ bias,   // [D]
    float*        __restrict__ out)    // [B,D]
{
    __shared__ __align__(16) SMem sm;

    const int tid  = threadIdx.x;
    const int wave = tid >> 5, lane = tid & 31;
    const int wm = wave & 3;          // 0..3  -> M offset wm*32
    const int wn = wave >> 2;         // 0..1  -> N offset wn*32
    const int lr = lane & 15, lh = lane >> 4;
    const int m0 = blockIdx.x * BM;
    const int n0 = blockIdx.y * BN;

    v8f acc[2][2] = {};

    for (int k0 = 0; k0 < D_DIM; k0 += BK) {
#if USE_TDM
        // ---- B tile via Tensor Data Mover (wave 0 issues one 2D-tile DMA) ----
        if (wave == 0) {
            unsigned long long ga =
                (unsigned long long)(const void*)&Wb[(size_t)n0 * D_DIM + k0];
            v4u g0;
            g0[0] = 1u;                                  // count=1, no gather
            g0[1] = ABYTES;                              // lds_addr = &sm.BsT
            g0[2] = (unsigned)ga;                        // global_addr lo
            g0[3] = (unsigned)((ga >> 32) & 0x01FFFFFFu) // global_addr hi (57b)
                    | (2u << 30);                        // type=2 ("image")
            v8i g1;
            g1[0] = (int)((1u << 16)     // data_size = 2 bytes
                        | (1u << 20)     // pad_enable
                        | (4u << 22)     // pad_interval: 32 DWORDs = 128B row
                        | (3u << 25));   // pad_amount:   4 DWORDs = 16B (=8 bf16)
            g1[1] = (int)(((unsigned)D_DIM & 0xFFFFu) << 16); // tensor_dim0 lo16
            g1[2] = (int)(((unsigned)D_DIM & 0xFFFF0000u) >> 16)
                  | (int)(((unsigned)D_DIM & 0xFFFFu) << 16); // dim0 hi16 | dim1 lo16
            g1[3] = (int)((unsigned)BK << 16);           // dim1 hi16=0 | tile_dim0
            g1[4] = BN;                                  // tile_dim1 | tile_dim2=0
            g1[5] = D_DIM;                               // tensor_dim0_stride lo32
            g1[6] = 0;                                   // stride hi | dim1_stride lo
            g1[7] = 0;
            v4i g2 = {0, 0, 0, 0};
            v4i g3 = {0, 0, 0, 0};
            v8i g4 = {0, 0, 0, 0, 0, 0, 0, 0};  // 6-arg toolchain: zero-filled
            __builtin_amdgcn_tensor_load_to_lds(g0, g1, g2, g3, g4, 0);
        }
#else
        // ---- fallback: wide manual staging using W symmetry ----
        for (int i = tid; i < (BN * BK) / 8; i += 256) {   // 512 x 16B copies
            int c = i >> 3, r8 = i & 7;
            *(v8bf*)&sm.BsT[c][r8 * 8] =
                *(const v8bf*)&Wb[(size_t)(n0 + c) * D_DIM + k0 + r8 * 8];
        }
#endif
        // ---- A tile: float4 loads, packed f32->bf16 convert, 8B LDS stores ----
        for (int i = tid; i < (BM * BK) / 4; i += 256) {   // 2048 float4
            int r = i >> 4, c4 = i & 15;                   // BK/4 = 16
            float4 f = *(const float4*)&X[(size_t)(m0 + r) * D_DIM + k0 + c4 * 4];
            v4bf pk = { (__bf16)f.x, (__bf16)f.y, (__bf16)f.z, (__bf16)f.w };
            *(v4bf*)&sm.As[r][c4 * 4] = pk;
        }
        if (k0 + BK < D_DIM)   // speculative prefetch of next X tile rows
            __builtin_prefetch(&X[(size_t)(m0 + (tid & 127)) * D_DIM + k0 + BK], 0, 1);
#if USE_TDM
        if (wave == 0) __builtin_amdgcn_s_wait_tensorcnt(0);
#endif
        __syncthreads();

        // ---- two WMMA K-substeps per stage ----
        #pragma unroll
        for (int kk = 0; kk < BK; kk += 32) {
            v16bf a[2], b[2];
            #pragma unroll
            for (int mi = 0; mi < 2; ++mi) {
                int row = wm * 32 + mi * 16 + lr;
                v8bf lo = *(const v8bf*)&sm.As[row][kk + lh * 8];
                v8bf hi = *(const v8bf*)&sm.As[row][kk + 16 + lh * 8];
                a[mi] = __builtin_shufflevector(lo, hi,
                        0,1,2,3,4,5,6,7,8,9,10,11,12,13,14,15);
            }
            #pragma unroll
            for (int ni = 0; ni < 2; ++ni) {
                int col = wn * 32 + ni * 16 + lr;
                v8bf lo = *(const v8bf*)&sm.BsT[col][kk + lh * 8];
                v8bf hi = *(const v8bf*)&sm.BsT[col][kk + 16 + lh * 8];
                b[ni] = __builtin_shufflevector(lo, hi,
                        0,1,2,3,4,5,6,7,8,9,10,11,12,13,14,15);
            }
            #pragma unroll
            for (int mi = 0; mi < 2; ++mi)
                #pragma unroll
                for (int ni = 0; ni < 2; ++ni)
                    acc[mi][ni] = __builtin_amdgcn_wmma_f32_16x16x32_bf16(
                        false, a[mi], false, b[ni], (short)0, acc[mi][ni],
                        false, false);
        }
        __syncthreads();
    }

    // epilogue: C/D layout -> VGPR r holds M=(lh?8:0)+r, N=lr; fuse +bias
    #pragma unroll
    for (int mi = 0; mi < 2; ++mi) {
        #pragma unroll
        for (int ni = 0; ni < 2; ++ni) {
            int mrow = m0 + wm * 32 + mi * 16 + lh * 8;
            int col  = n0 + wn * 32 + ni * 16 + lr;
            float bv = bias[col];
            #pragma unroll
            for (int r = 0; r < 8; ++r)
                out[(size_t)(mrow + r) * D_DIM + col] = acc[mi][ni][r] + bv;
        }
    }
}

// ---------------------------------------------------------------------------
// Kernel 3: per-row forward pass.  One block (256 threads) per batch row.
//   score_x = 0.5 * x'(grad + bias)   (since grad = Wx + b)
//   31 sequential categorical-without-replacement steps via Gumbel-max,
//   fusing the forward masked-log-softmax sum (same blocked-set evolution).
// ---------------------------------------------------------------------------
__global__ __launch_bounds__(256) void sample_fwd_kernel(
    const float* __restrict__ x, const float* __restrict__ grad,
    const float* __restrict__ bias,
    float* __restrict__ ybuf, int* __restrict__ idx_all,
    int* __restrict__ radius_buf, float* __restrict__ logfwd)
{
    constexpr int PT = D_DIM / 256;     // 16 elements per thread, strided
    const int row = blockIdx.x, tid = threadIdx.x;
    __shared__ float sred[256];
    __shared__ int   sidx[256];
    __shared__ float saux[256];
    __shared__ float sbc;

    float sc[PT], bits[PT];
    float dacc = 0.f;
    for (int e = 0; e < PT; ++e) {
        int i = e * 256 + tid;
        float xv = x[(size_t)row * D_DIM + i];
        float g  = grad[(size_t)row * D_DIM + i];
        dacc += xv * (g + bias[i]);
        sc[e]  = (1.f - 2.f * xv) * g * 0.5f;   // proposal logits
        bits[e] = xv;
    }
    sred[tid] = dacc; __syncthreads();
    for (int s = 128; s > 0; s >>= 1) { if (tid < s) sred[tid] += sred[tid + s]; __syncthreads(); }
    const float score_x = 0.5f * sred[0];
    __syncthreads();

    const int radius = 1 + (int)(pcg_hash(0x9e3779b9u ^ (unsigned)row) % 31u); // [1,31]
    unsigned blocked = 0;
    float gsum = 0.f;

    for (int j = 0; j < MAXR; ++j) {
        // local Gumbel-perturbed argmax + raw max over unblocked entries
        float pmax = NEG_BIG; int pidx = -1; float rmax = NEG_BIG;
        for (int e = 0; e < PT; ++e) {
            if (blocked & (1u << e)) continue;
            int i = e * 256 + tid;
            float u  = u01(pcg_hash(pcg_hash((unsigned)(row * MAXR + j)) ^ (unsigned)i));
            float pv = sc[e] - __logf(-__logf(u));
            if (pv > pmax) { pmax = pv; pidx = i; }
            rmax = fmaxf(rmax, sc[e]);
        }
        sred[tid] = pmax; sidx[tid] = pidx; saux[tid] = rmax; __syncthreads();
        for (int s = 128; s > 0; s >>= 1) {
            if (tid < s) {
                if (sred[tid + s] > sred[tid]) { sred[tid] = sred[tid + s]; sidx[tid] = sidx[tid + s]; }
                saux[tid] = fmaxf(saux[tid], saux[tid + s]);
            }
            __syncthreads();
        }
        const int idx = sidx[0]; const float mx = saux[0]; __syncthreads();

        // logsumexp over unblocked logits (for forward log-softmax term)
        float se = 0.f;
        for (int e = 0; e < PT; ++e)
            if (!(blocked & (1u << e))) se += __expf(sc[e] - mx);
        sred[tid] = se; __syncthreads();
        for (int s = 128; s > 0; s >>= 1) { if (tid < s) sred[tid] += sred[tid + s]; __syncthreads(); }
        const float lse = mx + __logf(sred[0]);
        __syncthreads();

        const int owner = idx & 255, slot = idx >> 8;
        if (tid == owner) {
            sbc = sc[slot];
            blocked |= (1u << slot);                        // blocked updates every step
            if (j < radius) bits[slot] = 1.f - bits[slot];  // flip only if masked-in
            idx_all[row * MAXR + j] = idx;
        }
        __syncthreads();
        if (j < radius) gsum += sbc - lse;
        __syncthreads();
    }

    for (int e = 0; e < PT; ++e)
        ybuf[(size_t)row * D_DIM + e * 256 + tid] = bits[e];
    if (tid == 0) { radius_buf[row] = radius; logfwd[row] = gsum + score_x; }
}

// ---------------------------------------------------------------------------
// Kernel 4: per-row backward pass + acceptance + final select.
//   Walk steps j = MAXR-1 .. 0; blocked set = indices of later steps ("after").
// ---------------------------------------------------------------------------
__global__ __launch_bounds__(256) void sample_bwd_kernel(
    const float* __restrict__ x, const float* __restrict__ ybuf,
    const float* __restrict__ grad, const float* __restrict__ bias,
    const int* __restrict__ idx_all, const int* __restrict__ radius_buf,
    const float* __restrict__ logfwd, float* __restrict__ out)
{
    constexpr int PT = D_DIM / 256;
    const int row = blockIdx.x, tid = threadIdx.x;
    __shared__ float sred[256];
    __shared__ float sbc;

    float sc[PT], ybits[PT];
    float dacc = 0.f;
    for (int e = 0; e < PT; ++e) {
        int i = e * 256 + tid;
        float yv = ybuf[(size_t)row * D_DIM + i];
        float g  = grad[(size_t)row * D_DIM + i];
        dacc += yv * (g + bias[i]);
        sc[e]  = (1.f - 2.f * yv) * g * 0.5f;
        ybits[e] = yv;
    }
    sred[tid] = dacc; __syncthreads();
    for (int s = 128; s > 0; s >>= 1) { if (tid < s) sred[tid] += sred[tid + s]; __syncthreads(); }
    const float score_y = 0.5f * sred[0];
    __syncthreads();

    const int radius = radius_buf[row];
    unsigned blocked = 0;
    float gsum = 0.f;

    for (int j = MAXR - 1; j >= 0; --j) {
        const int idx = idx_all[row * MAXR + j];

        float rmax = NEG_BIG;
        for (int e = 0; e < PT; ++e)
            if (!(blocked & (1u << e))) rmax = fmaxf(rmax, sc[e]);
        sred[tid] = rmax; __syncthreads();
        for (int s = 128; s > 0; s >>= 1) { if (tid < s) sred[tid] = fmaxf(sred[tid], sred[tid + s]); __syncthreads(); }
        const float mx = sred[0]; __syncthreads();

        float se = 0.f;
        for (int e = 0; e < PT; ++e)
            if (!(blocked & (1u << e))) se += __expf(sc[e] - mx);
        sred[tid] = se; __syncthreads();
        for (int s = 128; s > 0; s >>= 1) { if (tid < s) sred[tid] += sred[tid + s]; __syncthreads(); }
        const float lse = mx + __logf(sred[0]);
        __syncthreads();

        const int owner = idx & 255, slot = idx >> 8;
        if (tid == owner) sbc = sc[slot];
        __syncthreads();
        if (j < radius) gsum += sbc - lse;
        if (tid == owner) blocked |= (1u << slot);   // becomes "after" for j-1
        __syncthreads();
    }

    const float log_acc = (gsum + score_y) - logfwd[row];
    const float u = u01(pcg_hash(0xACCE55EDu ^ (unsigned)row));
    const float accept = (__expf(log_acc) >= u) ? 1.f : 0.f;

    for (int e = 0; e < PT; ++e) {
        int i = e * 256 + tid;
        float xv = x[(size_t)row * D_DIM + i];
        out[(size_t)row * D_DIM + i] = accept * ybits[e] + (1.f - accept) * xv;
    }
}

// ---------------------------------------------------------------------------
// Host launcher.  Workspace layout (~46.3 MB):
//   Wbf16 [D*D] | gradx [B*D] f32 | grady [B*D] f32 | y [B*D] f32
//   | idx_all [B*31] i32 | radius [B] i32 | logfwd [B] f32
// ---------------------------------------------------------------------------
extern "C" void kernel_launch(void* const* d_in, const int* in_sizes, int n_in,
                              void* d_out, int out_size, void* d_ws, size_t ws_size,
                              hipStream_t stream) {
    (void)in_sizes; (void)n_in; (void)out_size; (void)ws_size;

    const float* x    = (const float*)d_in[0];   // [B,D] 0/1 bits
    const float* W    = (const float*)d_in[1];   // [D,D] symmetric
    const float* bias = (const float*)d_in[2];   // [D]
    float*       out  = (float*)d_out;           // [B,D]

    char* ws = (char*)d_ws;
    size_t o = 0;
    __bf16* Wb     = (__bf16*)(ws + o); o += (size_t)D_DIM * D_DIM * sizeof(__bf16);
    float*  gradx  = (float*)(ws + o);  o += (size_t)B_DIM * D_DIM * sizeof(float);
    float*  grady  = (float*)(ws + o);  o += (size_t)B_DIM * D_DIM * sizeof(float);
    float*  ybuf   = (float*)(ws + o);  o += (size_t)B_DIM * D_DIM * sizeof(float);
    int*    idxall = (int*)(ws + o);    o += (size_t)B_DIM * MAXR * sizeof(int);
    int*    radius = (int*)(ws + o);    o += (size_t)B_DIM * sizeof(int);
    float*  logfwd = (float*)(ws + o);  o += (size_t)B_DIM * sizeof(float);

    dim3 ggrid(B_DIM / BM, D_DIM / BN);

    convert_w_kernel<<<D_DIM, 256, 0, stream>>>(W, Wb);
    gemm_wmma_kernel<<<ggrid, 256, 0, stream>>>(x, Wb, bias, gradx);
    sample_fwd_kernel<<<B_DIM, 256, 0, stream>>>(x, gradx, bias, ybuf, idxall, radius, logfwd);
    gemm_wmma_kernel<<<ggrid, 256, 0, stream>>>(ybuf, Wb, bias, grady);
    sample_bwd_kernel<<<B_DIM, 256, 0, stream>>>(x, ybuf, grady, bias, idxall, radius, logfwd, out);
}
